// Evo2Attention_6485400617087
// MI455X (gfx1250) — matrix operand order repaired and checked by reading, not compiled, not run on hardware
//
#include <hip/hip_runtime.h>
#include <hip/hip_bf16.h>

#define B_   2
#define S_   2048
#define HID_ 2048
#define NH_  16
#define HD_  128
#define BH_  (B_ * NH_)    // 32
#define M_   (B_ * S_)     // 4096

typedef __bf16 bf16_t;
typedef __attribute__((ext_vector_type(16))) __bf16 v16bf;
typedef __attribute__((ext_vector_type(8)))  __bf16 v8bf;
typedef __attribute__((ext_vector_type(8)))  float  v8f;

static __device__ __forceinline__ float bf2f(bf16_t x) {
  unsigned short h = __builtin_bit_cast(unsigned short, x);
  unsigned u = ((unsigned)h) << 16;
  return __builtin_bit_cast(float, u);
}
static __device__ __forceinline__ bf16_t f2bf(float f) {
  unsigned u = __builtin_bit_cast(unsigned, f);
  unsigned r = u + 0x7FFFu + ((u >> 16) & 1u);  // round-to-nearest-even
  unsigned short h = (unsigned short)(r >> 16);
  return __builtin_bit_cast(bf16_t, h);
}

static __device__ __forceinline__ v8f wmma_bf16(v16bf a, v16bf b, v8f c) {
  return __builtin_amdgcn_wmma_f32_16x16x32_bf16(
      /*neg_a=*/false, a, /*neg_b=*/false, b,
      /*c_mod=*/(short)0, c, /*reuse_a=*/false, /*reuse_b=*/false);
}

// A fragment: 16x32 bf16 tile (M x K), src row-major, base points at (row0,k0).
// Lanes 0-15: M=lane, K = 0..15; lanes 16-31: M=lane-16, K = 16..31 (contiguous).
static __device__ __forceinline__ v16bf load_afrag(const bf16_t* __restrict__ base,
                                                   int stride, int lane) {
  int m  = lane & 15;
  int kh = (lane >> 4) * 16;
  const bf16_t* p = base + (size_t)m * stride + kh;
  v8bf lo = *(const v8bf*)(p);
  v8bf hi = *(const v8bf*)(p + 8);
  v16bf r;
#pragma unroll
  for (int i = 0; i < 8; ++i) { r[i] = lo[i]; r[i + 8] = hi[i]; }
  return r;
}

// B fragment: 32x16 bf16 tile (K x N), sourced from an "N rows, K contiguous"
// buffer (i.e. we compute against W^T / K^T / V^T without transposing).
// Lane holds column n = lane&15; K-range = (lane>>4)*16 .. +15, contiguous.
static __device__ __forceinline__ v16bf load_bfrag(const bf16_t* __restrict__ base,
                                                   int stride, int lane) {
  int n  = lane & 15;
  int kh = (lane >> 4) * 16;
  return *(const v16bf*)(base + (size_t)n * stride + kh);
}

// ---------------------------------------------------------------------------
// fp32 -> bf16 convert
// ---------------------------------------------------------------------------
__global__ void f32_to_bf16_kernel(const float* __restrict__ in,
                                   bf16_t* __restrict__ out, size_t n) {
  size_t i = (size_t)blockIdx.x * blockDim.x + threadIdx.x;
  size_t step = (size_t)gridDim.x * blockDim.x;
  for (; i < n; i += step) out[i] = f2bf(in[i]);
}

// ---------------------------------------------------------------------------
// BF16 WMMA GEMM:  C[m,n] = sum_k A[m,k] * W[n,k]   (A: MxK rm, W: NxK rm)
// mode 0: out_b[((b*NH+h)*S + s)*HD + hd]   (Q/K layout [bh,s,hd])
// mode 2: out_b[((b*NH+h)*HD + hd)*S + s]   (V transposed layout [bh,hd,s])
// mode 1: out_f[m*N + n]                    (final fp32 output)
// Block: 128 thr (4 waves, 2x2), block tile 128x128, wave tile 64x64.
// 512 B of fragment traffic per WMMA; 16-deep WMMA chain per k-step.
// ---------------------------------------------------------------------------
__global__ __launch_bounds__(128) void gemm_bf16_kernel(
    const bf16_t* __restrict__ A, const bf16_t* __restrict__ W,
    float* __restrict__ outF, bf16_t* __restrict__ outB,
    int M, int N, int K, int mode) {
  int lane = threadIdx.x & 31;
  int wave = threadIdx.x >> 5;
  int m0 = blockIdx.y * 128 + (wave >> 1) * 64;
  int n0 = blockIdx.x * 128 + (wave & 1) * 64;

  v8f acc[4][4] = {};
  for (int k0 = 0; k0 < K; k0 += 32) {
    // prefetch two k-steps ahead into near caches (WGP-scope global_prefetch_b8)
    if (k0 + 64 < K) {
      int pr = (lane & 15) * 4;
      const bf16_t* pp = (lane < 16) ? (A + (size_t)(m0 + pr) * K + k0 + 64)
                                     : (W + (size_t)(n0 + pr) * K + k0 + 64);
      __builtin_prefetch(pp, 0, 3);
    }
    v16bf bfr[4];
#pragma unroll
    for (int j = 0; j < 4; ++j)
      bfr[j] = load_bfrag(W + (size_t)(n0 + j * 16) * K + k0, K, lane);
#pragma unroll
    for (int i = 0; i < 4; ++i) {
      v16bf af = load_afrag(A + (size_t)(m0 + i * 16) * K + k0, K, lane);
#pragma unroll
      for (int j = 0; j < 4; ++j) acc[i][j] = wmma_bf16(af, bfr[j], acc[i][j]);
    }
  }

  int nloc  = lane & 15;
  int mbase = (lane >> 4) * 8;
#pragma unroll
  for (int i = 0; i < 4; ++i) {
#pragma unroll
    for (int j = 0; j < 4; ++j) {
#pragma unroll
      for (int v = 0; v < 8; ++v) {
        int m = m0 + i * 16 + mbase + v;
        int n = n0 + j * 16 + nloc;
        float val = acc[i][j][v];
        if (mode == 1) {
          outF[(size_t)m * N + n] = val;
        } else {
          int b = m >> 11, s = m & (S_ - 1);
          int h = n >> 7,  hd = n & (HD_ - 1);
          size_t idx = (mode == 0)
              ? (((size_t)b * NH_ + h) * S_ + s) * HD_ + hd
              : (((size_t)b * NH_ + h) * HD_ + hd) * S_ + s;
          outB[idx] = f2bf(val);
        }
      }
    }
  }
}

// ---------------------------------------------------------------------------
// RoPE in-place on [BH, S, HD] bf16; thread = (bh, s, pair i in 0..63)
// angle = pos * 10000^(-((2i) mod 64)/64)
// ---------------------------------------------------------------------------
__global__ void rope_kernel(bf16_t* __restrict__ X) {
  size_t idx = (size_t)blockIdx.x * blockDim.x + threadIdx.x;
  if (idx >= (size_t)BH_ * S_ * (HD_ / 2)) return;
  int i = (int)(idx & 63);
  size_t row = idx >> 6;             // bh*S + s
  int pos = (int)(row & (S_ - 1));
  int j2 = (2 * i) & 63;
  float freq = __expf(-(float)j2 * (9.210340371976184f / 64.0f)); // ln(1e4)
  float th = (float)pos * freq;
  float sv, cv;
  sincosf(th, &sv, &cv);
  bf16_t* p = X + row * HD_ + 2 * i;
  float x1 = bf2f(p[0]), x2 = bf2f(p[1]);
  p[0] = f2bf(x1 * cv - x2 * sv);
  p[1] = f2bf(x1 * sv + x2 * cv);
}

// ---------------------------------------------------------------------------
// Flash attention (causal). Grid: (S/64, BH). Block: 128 thr = 4 waves.
// Each wave owns 16 query rows; k-tiles of 32 keys; online softmax via
// per-wave LDS slices (wave-lockstep, no __syncthreads needed).
// Q,K: [bh,s,hd] bf16; V: [bh,hd,s] bf16; O: [b,s,hid] bf16.
// ---------------------------------------------------------------------------
__global__ __launch_bounds__(128) void flash_attn_kernel(
    const bf16_t* __restrict__ Qb, const bf16_t* __restrict__ Kb,
    const bf16_t* __restrict__ Vt, bf16_t* __restrict__ Ob) {
  __shared__ __align__(32) float  s_sc[4][16][32];
  __shared__ __align__(32) bf16_t s_p[4][16][32];
  __shared__ float s_m[4][16], s_l[4][16], s_alpha[4][16];

  int lane = threadIdx.x & 31;
  int wave = threadIdx.x >> 5;
  int bh = blockIdx.y;
  int q0 = blockIdx.x * 64 + wave * 16;

  const bf16_t* Qp = Qb + ((size_t)bh * S_ + q0) * HD_;
  const bf16_t* Kp = Kb + (size_t)bh * S_ * HD_;
  const bf16_t* Vp = Vt + (size_t)bh * HD_ * S_;

  v16bf qf[4];
#pragma unroll
  for (int c = 0; c < 4; ++c) qf[c] = load_afrag(Qp + c * 32, HD_, lane);

  v8f oacc[8] = {};
  if (lane < 16) { s_m[wave][lane] = -1e30f; s_l[wave][lane] = 0.0f; }
  __builtin_amdgcn_wave_barrier();

  const float scale = 0.08838834764831845f;  // 1/sqrt(128)
  int nloc  = lane & 15;
  int mbase = (lane >> 4) * 8;
  int r    = lane >> 1;
  int half = lane & 1;
  int qrow = q0 + r;

  for (int k0 = 0; k0 < q0 + 16; k0 += 32) {
    // ---- scores: S = Q (16x128) @ K^T (128x32), two 16x16 n-tiles ----
    v8f sc[2] = {};
#pragma unroll
    for (int j = 0; j < 2; ++j)
#pragma unroll
      for (int c = 0; c < 4; ++c) {
        v16bf kf = load_bfrag(Kp + (size_t)(k0 + j * 16) * HD_ + c * 32, HD_, lane);
        sc[j] = wmma_bf16(qf[c], kf, sc[j]);
      }
#pragma unroll
    for (int j = 0; j < 2; ++j)
#pragma unroll
      for (int v = 0; v < 8; ++v)
        s_sc[wave][mbase + v][j * 16 + nloc] = sc[j][v];
    __builtin_amdgcn_wave_barrier();

    // ---- online softmax: 2 lanes per row, 16 keys each ----
    float vals[16];
    float pm = -1e30f;
#pragma unroll
    for (int c = 0; c < 16; ++c) {
      int key = k0 + half * 16 + c;
      float sv = s_sc[wave][r][half * 16 + c] * scale;
      if (key > qrow) sv = -1e30f;
      vals[c] = sv;
      pm = fmaxf(pm, sv);
    }
    float tmax = fmaxf(pm, __shfl_xor(pm, 1));
    float mold = s_m[wave][r];
    float mnew = fmaxf(mold, tmax);
    float psum = 0.0f;
#pragma unroll
    for (int c = 0; c < 16; ++c) {
      float p = __expf(vals[c] - mnew);
      psum += p;
      s_p[wave][r][half * 16 + c] = f2bf(p);
    }
    float tsum = psum + __shfl_xor(psum, 1);
    float alpha = __expf(mold - mnew);
    if (half == 0) {
      s_alpha[wave][r] = alpha;
      s_m[wave][r] = mnew;
      s_l[wave][r] = s_l[wave][r] * alpha + tsum;
    }
    __builtin_amdgcn_wave_barrier();

    // ---- rescale accumulator, then O += P (16x32) @ V (32x128) ----
    float al[8];
#pragma unroll
    for (int v = 0; v < 8; ++v) al[v] = s_alpha[wave][mbase + v];
#pragma unroll
    for (int t = 0; t < 8; ++t)
#pragma unroll
      for (int v = 0; v < 8; ++v) oacc[t][v] *= al[v];

    v16bf pf = load_afrag(&s_p[wave][0][0], 32, lane);
#pragma unroll
    for (int t = 0; t < 8; ++t) {
      v16bf vf = load_bfrag(Vp + (size_t)(t * 16) * S_ + k0, S_, lane);
      oacc[t] = wmma_bf16(pf, vf, oacc[t]);
    }
  }

  __builtin_amdgcn_wave_barrier();
  float li[8];
#pragma unroll
  for (int v = 0; v < 8; ++v) li[v] = 1.0f / s_l[wave][mbase + v];

  int b = bh / NH_, h = bh % NH_;
#pragma unroll
  for (int t = 0; t < 8; ++t)
#pragma unroll
    for (int v = 0; v < 8; ++v) {
      int srow = q0 + mbase + v;
      int hd = t * 16 + nloc;
      Ob[((size_t)b * S_ + srow) * HID_ + h * HD_ + hd] = f2bf(oacc[t][v] * li[v]);
    }
}

// ---------------------------------------------------------------------------
// host launcher
// ---------------------------------------------------------------------------
extern "C" void kernel_launch(void* const* d_in, const int* in_sizes, int n_in,
                              void* d_out, int out_size, void* d_ws, size_t ws_size,
                              hipStream_t stream) {
  const float* hs = (const float*)d_in[0];
  const float* Wq = (const float*)d_in[1];
  const float* Wk = (const float*)d_in[2];
  const float* Wv = (const float*)d_in[3];
  const float* Wo = (const float*)d_in[4];
  (void)in_sizes; (void)n_in; (void)out_size; (void)ws_size;

  bf16_t* ws = (bf16_t*)d_ws;
  size_t o = 0;
  const size_t NXE = (size_t)M_ * HID_;     // 8,388,608
  const size_t NWE = (size_t)HID_ * HID_;   // 4,194,304
  bf16_t* Xb  = ws + o; o += NXE;
  bf16_t* Wqb = ws + o; o += NWE;
  bf16_t* Wkb = ws + o; o += NWE;
  bf16_t* Wvb = ws + o; o += NWE;
  bf16_t* Wob = ws + o; o += NWE;
  bf16_t* Qb  = ws + o; o += NXE;
  bf16_t* Kb  = ws + o; o += NXE;
  bf16_t* Vt  = ws + o; o += NXE;
  bf16_t* Ob  = ws + o; o += NXE;

  // 1) fp32 -> bf16 conversions
  f32_to_bf16_kernel<<<4096, 256, 0, stream>>>(hs, Xb, NXE);
  f32_to_bf16_kernel<<<4096, 256, 0, stream>>>(Wq, Wqb, NWE);
  f32_to_bf16_kernel<<<4096, 256, 0, stream>>>(Wk, Wkb, NWE);
  f32_to_bf16_kernel<<<4096, 256, 0, stream>>>(Wv, Wvb, NWE);
  f32_to_bf16_kernel<<<4096, 256, 0, stream>>>(Wo, Wob, NWE);

  // 2) QKV projections (V stored transposed per head)
  dim3 gb(128);
  dim3 gg(HID_ / 128, M_ / 128);
  gemm_bf16_kernel<<<gg, gb, 0, stream>>>(Xb, Wqb, nullptr, Qb, M_, HID_, HID_, 0);
  gemm_bf16_kernel<<<gg, gb, 0, stream>>>(Xb, Wkb, nullptr, Kb, M_, HID_, HID_, 0);
  gemm_bf16_kernel<<<gg, gb, 0, stream>>>(Xb, Wvb, nullptr, Vt, M_, HID_, HID_, 2);

  // 3) RoPE on Q and K
  int rope_total = BH_ * S_ * (HD_ / 2);
  rope_kernel<<<(rope_total + 255) / 256, 256, 0, stream>>>(Qb);
  rope_kernel<<<(rope_total + 255) / 256, 256, 0, stream>>>(Kb);

  // 4) causal flash attention
  flash_attn_kernel<<<dim3(S_ / 64, BH_), 128, 0, stream>>>(Qb, Kb, Vt, Ob);

  // 5) output projection -> fp32 d_out
  gemm_bf16_kernel<<<gg, gb, 0, stream>>>(Ob, Wob, (float*)d_out, nullptr,
                                          M_, HID_, HID_, 1);
}
